// _CONVERSE_single_16879221473979
// MI455X (gfx1250) — matrix operand order repaired and checked by reading, not compiled, not run on hardware
//
#include <hip/hip_runtime.h>
#include <math.h>

typedef __attribute__((ext_vector_type(16))) _Float16 v16h;
typedef __attribute__((ext_vector_type(8)))  _Float16 v8h;
typedef __attribute__((ext_vector_type(8)))  float    v8f;

#define IN_DIM  128
#define LDIM    16
#define KCL     8
#define DDUR    50
#define ENC_H   64
#define DEC_H   16
#define HEAD_H  64
#define HF      (LDIM + IN_DIM)   /* 144 */
#define HFP     160               /* padded to 5 K-chunks of 32 */
#define WAVES_PER_BLOCK 8
#define TILES_PER_WAVE  4
#define ROWS_PER_TILE   16
#define ROWS_PER_BLOCK  (WAVES_PER_BLOCK * TILES_PER_WAVE * ROWS_PER_TILE) /* 512 */

/* weight-fragment directory (units: fragments of 512 f16 = 32 lanes x 16 halves) */
#define OFF_E1  0                 /* We1: 4 kc x 4 nt = 16  */
#define OFF_E2  16                /* We2: 2 x 4       = 8   */
#define OFF_E3  24                /* We3: 2 x 1       = 2   */
#define OFF_H1  26                /* Wh1: 8 heads x (5x4) = 160 */
#define OFF_H2  186               /* Wh2: 8 heads x (2x4) = 64  */
#define OFF_D1  250               /* Wd1: 1 */
#define OFF_D2  251               /* Wd2: 1 x 8 = 8 */
#define NFRAG   259

__device__ __forceinline__ v8f wmma16(v16h a, v16h b, v8f c) {
  return __builtin_amdgcn_wmma_f32_16x16x32_f16(
      false, a, false, b, (short)0, c, false, false);
}

__device__ __forceinline__ v16h combine16(v8h lo, v8h hi) {
  return __builtin_shufflevector(lo, hi, 0, 1, 2, 3, 4, 5, 6, 7,
                                         8, 9, 10, 11, 12, 13, 14, 15);
}

// A fragment (16x32 f16) from LDS row-major [16][ld]; both K-runs are contiguous
// 16-byte chunks => two ds_load_b128 per lane.
__device__ __forceinline__ v16h load_a16(const _Float16* base, int ld, int k0, int lane) {
  const int row = lane & 15;
  const int kb  = k0 + (lane >> 4) * 8;
  const _Float16* p = base + row * ld + kb;
  v8h lo = *(const v8h*)(p);
  v8h hi = *(const v8h*)(p + 16);
  return combine16(lo, hi);
}

// A fragment with only K=0..15 valid (K padded to 32 with zeros): decoder path.
__device__ __forceinline__ v16h load_a_lo(const _Float16* base, int ld, int lane) {
  const int row = lane & 15;
  const int kb  = (lane >> 4) * 8;
  v8h lo = *(const v8h*)(base + row * ld + kb);
  v8h hi = {};
  return combine16(lo, hi);
}

// B fragment from pre-swizzled LDS: 16 contiguous halves per lane.
__device__ __forceinline__ v16h load_b_frag(const _Float16* wf, int fragIdx, int lane) {
  const _Float16* p = wf + fragIdx * 512 + lane * 16;
  v8h lo = *(const v8h*)(p);
  v8h hi = *(const v8h*)(p + 8);
  return combine16(lo, hi);
}

// Cooperative (256-thread) staging of a global f32 matrix into f16 B-fragment
// layout in LDS. Reads are row-major coalesced; OOB (padding) rows/cols -> 0.
__device__ __forceinline__ void stage_weights(_Float16* dst, const float* __restrict__ W,
                                              int ldw, int K, int Nn,
                                              int kcCount, int ntCount, int tid) {
  const int Npad  = ntCount * 16;          // power of two at every call site
  const int total = kcCount * 32 * Npad;
  for (int e = tid; e < total; e += 256) {
    int   k   = e / Npad;
    int   col = e - k * Npad;
    float v   = (k < K && col < Nn) ? W[k * ldw + col] : 0.0f;
    int kc = k >> 5, kr = k & 31;
    int nt = col >> 4, nc = col & 15;
    int l  = (((kr >> 3) & 1) << 4) | nc;  // lane
    int h  = (kr & 7) | ((kr >> 4) << 3);  // half index within lane
    dst[(kc * ntCount + nt) * 512 + l * 16 + h] = (_Float16)v;
  }
}

__global__ void __launch_bounds__(256)
fused_dcsurv_kernel(const float* __restrict__ x,
                    const float* __restrict__ We1, const float* __restrict__ be1,
                    const float* __restrict__ We2, const float* __restrict__ be2,
                    const float* __restrict__ We3, const float* __restrict__ be3,
                    const float* __restrict__ Wd1, const float* __restrict__ bd1,
                    const float* __restrict__ Wd2, const float* __restrict__ bd2,
                    const float* __restrict__ Wh1, const float* __restrict__ bh1,
                    const float* __restrict__ Wh2, const float* __restrict__ bh2,
                    const float* __restrict__ centers,
                    float* __restrict__ out_z,  float* __restrict__ out_q,
                    float* __restrict__ out_s,  float* __restrict__ out_x,
                    float* __restrict__ out_rec, float* __restrict__ out_kld,
                    int N)
{
  // 265,216 + 40,960 + 16,384 + 512 = 323,072 B  (<= 320 KB WGP LDS)
  __shared__ __align__(16) _Float16 wf_s[NFRAG * 512];
  __shared__ __align__(16) _Float16 hs_s[WAVES_PER_BLOCK][ROWS_PER_TILE * HFP];
  __shared__ __align__(16) _Float16 buf_s[WAVES_PER_BLOCK][ROWS_PER_TILE * 64];
  __shared__ int idx_s[WAVES_PER_BLOCK][ROWS_PER_TILE];

  const int tid  = threadIdx.x;
  const int lane = tid & 31;
  const int wave = tid >> 5;
  const int base = blockIdx.x * ROWS_PER_BLOCK;

  // ---------------- cooperative weight staging (once per block) ----------------
  stage_weights(wf_s + OFF_E1 * 512, We1, ENC_H, IN_DIM, ENC_H, 4, 4, tid);
  stage_weights(wf_s + OFF_E2 * 512, We2, ENC_H, ENC_H, ENC_H, 2, 4, tid);
  stage_weights(wf_s + OFF_E3 * 512, We3, LDIM, ENC_H, LDIM, 2, 1, tid);
  for (int k = 0; k < KCL; ++k) {
    stage_weights(wf_s + (OFF_H1 + k * 20) * 512, Wh1 + (size_t)k * HF * HEAD_H,
                  HEAD_H, HF, HEAD_H, 5, 4, tid);
    stage_weights(wf_s + (OFF_H2 + k * 8) * 512, Wh2 + (size_t)k * HEAD_H * DDUR,
                  DDUR, HEAD_H, DDUR, 2, 4, tid);
  }
  stage_weights(wf_s + OFF_D1 * 512, Wd1, DEC_H, LDIM, DEC_H, 1, 1, tid);
  stage_weights(wf_s + OFF_D2 * 512, Wd2, IN_DIM, DEC_H, IN_DIM, 1, 8, tid);
  __syncthreads();

  _Float16* hs  = hs_s[wave];
  _Float16* buf = buf_s[wave];
  int*      idx = idx_s[wave];

  const int colb  = lane & 15;
  const int rbase = (lane >> 4) * 8;

  for (int t = 0; t < TILES_PER_WAVE; ++t) {
    const int m0 = base + (t * WAVES_PER_BLOCK + wave) * ROWS_PER_TILE;
    if (m0 >= N) break; // wave-uniform

    // prefetch next tile's x rows (8 KB = 64 x 128B lines; 2 wave prefetches)
    if (t + 1 < TILES_PER_WAVE) {
      const int m0n = base + ((t + 1) * WAVES_PER_BLOCK + wave) * ROWS_PER_TILE;
      if (m0n < N) {
        const char* p = (const char*)(x + (size_t)m0n * IN_DIM);
        __builtin_prefetch(p + (size_t)lane * 128, 0, 1);
        __builtin_prefetch(p + (size_t)(lane + 32) * 128, 0, 1);
      }
    }

    // -------------- stage h = [z(16) | x(128) | pad(16)] as f16 ---------------
    for (int r = 0; r < ROWS_PER_TILE; ++r) {
#pragma unroll
      for (int j = 0; j < HFP / 32; ++j) {
        int   c = j * 32 + lane;
        float v = (c >= LDIM && c < LDIM + IN_DIM)
                    ? x[(size_t)(m0 + r) * IN_DIM + (c - LDIM)] : 0.0f;
        hs[r * HFP + c] = (_Float16)v;
      }
    }

    // -------------- encoder layer 1: e1 = relu(x @ We1 + be1)  K=128 ----------
    {
      v8f acc[4] = {};
#pragma unroll
      for (int kc = 0; kc < 4; ++kc) {
        v16h a = load_a16(hs, HFP, LDIM + kc * 32, lane);
#pragma unroll
        for (int nt = 0; nt < 4; ++nt) {
          v16h b = load_b_frag(wf_s, OFF_E1 + kc * 4 + nt, lane);
          acc[nt] = wmma16(a, b, acc[nt]);
        }
      }
#pragma unroll
      for (int nt = 0; nt < 4; ++nt) {
        float bias = be1[nt * 16 + colb];
#pragma unroll
        for (int v = 0; v < 8; ++v) {
          float e = fmaxf(acc[nt][v] + bias, 0.0f);
          buf[(rbase + v) * 64 + nt * 16 + colb] = (_Float16)e;
        }
      }
    }

    // -------------- encoder layer 2: e2 = relu(e1 @ We2 + be2)  K=64 ----------
    {
      v8f acc[4] = {};
#pragma unroll
      for (int kc = 0; kc < 2; ++kc) {
        v16h a = load_a16(buf, 64, kc * 32, lane);
#pragma unroll
        for (int nt = 0; nt < 4; ++nt) {
          v16h b = load_b_frag(wf_s, OFF_E2 + kc * 4 + nt, lane);
          acc[nt] = wmma16(a, b, acc[nt]);
        }
      }
#pragma unroll
      for (int nt = 0; nt < 4; ++nt) {
        float bias = be2[nt * 16 + colb];
#pragma unroll
        for (int v = 0; v < 8; ++v) {
          float e = fmaxf(acc[nt][v] + bias, 0.0f);
          buf[(rbase + v) * 64 + nt * 16 + colb] = (_Float16)e;
        }
      }
    }

    // -------------- encoder layer 3: z = e2 @ We3 + be3  [16,16] --------------
    {
      v8f zacc = {};
#pragma unroll
      for (int kc = 0; kc < 2; ++kc) {
        v16h a = load_a16(buf, 64, kc * 32, lane);
        v16h b = load_b_frag(wf_s, OFF_E3 + kc, lane);
        zacc = wmma16(a, b, zacc);
      }
      float bias = be3[colb];
#pragma unroll
      for (int v = 0; v < 8; ++v) {
        int   row = rbase + v;
        float zv  = zacc[v] + bias;
        out_z[(size_t)(m0 + row) * LDIM + colb] = zv;
        hs[row * HFP + colb] = (_Float16)zv;
      }
    }

    // -------------- student-t q + softmax + argmax (lanes 0..15) --------------
    if (lane < ROWS_PER_TILE) {
      int   row = lane;
      float zv[LDIM];
#pragma unroll
      for (int d = 0; d < LDIM; ++d) zv[d] = (float)hs[row * HFP + d];
      float lgk[KCL];
      float mx = -3.402823466e38f;
#pragma unroll
      for (int k = 0; k < KCL; ++k) {
        float d2 = 0.0f;
#pragma unroll
        for (int d = 0; d < LDIM; ++d) {
          float df = zv[d] - centers[k * LDIM + d];
          d2 += df * df;
        }
        lgk[k] = -log1pf(d2);            // -0.5*(DF+1)*log1p(d2/DF), DF=1
        mx = fmaxf(mx, lgk[k]);
      }
      float ex[KCL]; float s = 0.0f;
#pragma unroll
      for (int k = 0; k < KCL; ++k) { ex[k] = __expf(lgk[k] - mx); s += ex[k]; }
      float inv = 1.0f / s;
      int best = 0; float bv = lgk[0];
#pragma unroll
      for (int k = 1; k < KCL; ++k) { if (lgk[k] > bv) { bv = lgk[k]; best = k; } }
#pragma unroll
      for (int k = 0; k < KCL; ++k) out_q[(size_t)(m0 + row) * KCL + k] = ex[k] * inv;
      idx[row] = best;
    }

    // -------------- K per-cluster survival heads ------------------------------
    for (int k = 0; k < KCL; ++k) {
      const float* bh1k = bh1 + k * HEAD_H;
      const float* bh2k = bh2 + k * DDUR;

      v8f h1[4] = {};
#pragma unroll
      for (int kc = 0; kc < 5; ++kc) {
        v16h a = load_a16(hs, HFP, kc * 32, lane);
#pragma unroll
        for (int nt = 0; nt < 4; ++nt) {
          v16h b = load_b_frag(wf_s, OFF_H1 + k * 20 + kc * 4 + nt, lane);
          h1[nt] = wmma16(a, b, h1[nt]);
        }
      }
#pragma unroll
      for (int nt = 0; nt < 4; ++nt) {
        float bias = bh1k[nt * 16 + colb];
#pragma unroll
        for (int v = 0; v < 8; ++v) {
          float e = fmaxf(h1[nt][v] + bias, 0.0f);
          buf[(rbase + v) * 64 + nt * 16 + colb] = (_Float16)e;
        }
      }

      v8f lg[4] = {};
#pragma unroll
      for (int kc = 0; kc < 2; ++kc) {
        v16h a = load_a16(buf, 64, kc * 32, lane);
#pragma unroll
        for (int nt = 0; nt < 4; ++nt) {
          v16h b = load_b_frag(wf_s, OFF_H2 + k * 8 + kc * 4 + nt, lane);
          lg[nt] = wmma16(a, b, lg[nt]);
        }
      }
#pragma unroll
      for (int nt = 0; nt < 4; ++nt) {
        int col = nt * 16 + colb;
        if (col < DDUR) {
          float bias = bh2k[col];
#pragma unroll
          for (int v = 0; v < 8; ++v) {
            int row = rbase + v;
            if (idx[row] == k)
              out_s[(size_t)(m0 + row) * DDUR + col] = lg[nt][v] + bias;
          }
        }
      }
    }

    // -------------- decoder + per-sample reconstruction loss ------------------
    {
      v16h az = load_a_lo(hs, HFP, lane);               // z, K=16 padded to 32
      v16h b1 = load_b_frag(wf_s, OFF_D1, lane);
      v8f  cd = {};
      cd = wmma16(az, b1, cd);
      float bias1 = bd1[colb];
#pragma unroll
      for (int v = 0; v < 8; ++v) {
        float e = fmaxf(cd[v] + bias1, 0.0f);
        buf[(rbase + v) * 64 + colb] = (_Float16)e;     // d in cols 0..15
      }

      v16h ad = load_a_lo(buf, 64, lane);               // d, K=16 padded to 32
      float racc[8];
#pragma unroll
      for (int v = 0; v < 8; ++v) racc[v] = 0.0f;

#pragma unroll
      for (int nt = 0; nt < 8; ++nt) {
        v16h bw = load_b_frag(wf_s, OFF_D2 + nt, lane);
        v8f  cx = {};
        cx = wmma16(ad, bw, cx);
        int   col   = nt * 16 + colb;
        float bias2 = bd2[col];
#pragma unroll
        for (int v = 0; v < 8; ++v) {
          int   row = rbase + v;
          float xh  = cx[v] + bias2;
          out_x[(size_t)(m0 + row) * IN_DIM + col] = xh;
          float xv = (float)hs[row * HFP + LDIM + col];
          float df = xh - xv;
          racc[v] += df * df;
        }
      }

#pragma unroll
      for (int off = 1; off < 16; off <<= 1) {
#pragma unroll
        for (int v = 0; v < 8; ++v)
          racc[v] += __shfl_xor(racc[v], off, 32);
      }
      if ((lane & 15) == 0) {
#pragma unroll
        for (int v = 0; v < 8; ++v) {
          int row = rbase + v;
          out_rec[m0 + row] = racc[v] * (1.0f / IN_DIM);
          out_kld[m0 + row] = 0.0f;
        }
      }
    }
  } // tile loop
}

extern "C" void kernel_launch(void* const* d_in, const int* in_sizes, int n_in,
                              void* d_out, int out_size, void* d_ws, size_t ws_size,
                              hipStream_t stream) {
  const float* x       = (const float*)d_in[0];
  const float* We1     = (const float*)d_in[1];
  const float* be1     = (const float*)d_in[2];
  const float* We2     = (const float*)d_in[3];
  const float* be2     = (const float*)d_in[4];
  const float* We3     = (const float*)d_in[5];
  const float* be3     = (const float*)d_in[6];
  const float* Wd1     = (const float*)d_in[7];
  const float* bd1     = (const float*)d_in[8];
  const float* Wd2     = (const float*)d_in[9];
  const float* bd2     = (const float*)d_in[10];
  const float* Wh1     = (const float*)d_in[11];
  const float* bh1     = (const float*)d_in[12];
  const float* Wh2     = (const float*)d_in[13];
  const float* bh2     = (const float*)d_in[14];
  const float* centers = (const float*)d_in[15];

  const int N = in_sizes[0] / IN_DIM;

  float* out_z   = (float*)d_out;
  float* out_q   = out_z   + (size_t)N * LDIM;
  float* out_s   = out_q   + (size_t)N * KCL;
  float* out_x   = out_s   + (size_t)N * DDUR;
  float* out_rec = out_x   + (size_t)N * IN_DIM;
  float* out_kld = out_rec + (size_t)N;

  const int grid = (N + ROWS_PER_BLOCK - 1) / ROWS_PER_BLOCK; // 512 samples/block

  fused_dcsurv_kernel<<<grid, 256, 0, stream>>>(
      x, We1, be1, We2, be2, We3, be3, Wd1, bd1, Wd2, bd2,
      Wh1, bh1, Wh2, bh2, centers,
      out_z, out_q, out_s, out_x, out_rec, out_kld, N);
}
